// Demo_79525614452886
// MI455X (gfx1250) — matrix operand order repaired and checked by reading, not compiled, not run on hardware
//
#include <hip/hip_runtime.h>

// ---------------------------------------------------------------------------
// W4A16 GEMV for MI455X (gfx1250), M=1, K=8192, N=16384, GROUP=32.
//
// Roofline: B = 256MB (int32-per-packed-byte), scales = 16MB -> ~11.7us at
// 23.3 TB/s. Memory bound. WMMA f16 (16x16x32, K == GROUP) eats all MACs so
// VALU budget is spent purely on int4->fp16 dequant (~1.25 ops/byte).
// ---------------------------------------------------------------------------

typedef __attribute__((ext_vector_type(16))) _Float16     v16h;
typedef __attribute__((ext_vector_type(2)))  _Float16     h2;
typedef __attribute__((ext_vector_type(8)))  float        v8f;
typedef __attribute__((ext_vector_type(8)))  unsigned int v8u;

#define KDIM        8192
#define GROUP       32
#define NGROUP      (KDIM / GROUP)        // 256
#define KB          (KDIM / 2)            // 4096 packed int32 per B row
#define WAVES       8
#define GPW         (NGROUP / WAVES)      // 32 groups per wave
#define PFDIST      8                     // prefetch distance in groups

union V16 {
    v8u  u;
    v16h h;
    uint4 q[2];
};

__global__ __launch_bounds__(256) void w4a16_gemv_wmma(
    const _Float16*     __restrict__ A,    // [1, K] fp16
    const int*          __restrict__ B,    // [N, K/2] int32, one packed byte each
    const unsigned int* __restrict__ SZ,   // [N, NGROUP] dword = {scale.f16, zero.f16}
    float*              __restrict__ out)  // [N]
{
    __shared__ _Float16 lA[KDIM];          // 16 KB staged activations
    __shared__ float    asum[NGROUP];      // per-group sum of A
    __shared__ float    partial[WAVES * 16];

    const int t = threadIdx.x;

    // ---- stage A into LDS (256 threads x 4 x 16B = 16KB) ----
    {
        const uint4* src = (const uint4*)A;
        uint4*       dst = (uint4*)lA;
        #pragma unroll
        for (int i = 0; i < 4; ++i)
            dst[t + 256 * i] = src[t + 256 * i];
    }
    __syncthreads();

    // ---- per-group activation sums (one group per thread) ----
    {
        float s = 0.f;
        const _Float16* ga = &lA[t * GROUP];
        #pragma unroll
        for (int i = 0; i < GROUP; ++i) s += (float)ga[i];
        asum[t] = s;
    }
    __syncthreads();

    const int wave  = t >> 5;              // wave32
    const int lane  = t & 31;
    const int col   = lane & 15;           // N within the 16-wide tile
    const int hi    = lane >> 4;           // 0: K 0-15, 1: K 16-31 of the group
    const int nbase = (int)blockIdx.x * 16;
    const int n     = nbase + col;

    const h2  bias  = { (_Float16)(-1032.0f), (_Float16)(-1032.0f) };
    const v8f czero = {};
    float acc = 0.f;

    const int  g0       = wave * GPW;
    const int* browbase = B + (long)n * KB;

    for (int gi = 0; gi < GPW; ++gi) {
        const int g   = g0 + gi;
        const int kb0 = g * 16 + hi * 8;   // 8 consecutive packed dwords per lane

        if (gi + PFDIST < GPW)
            __builtin_prefetch(browbase + kb0 + PFDIST * 16, 0, 0);

        // B-operand (contiguous k order, same as A: lanes 0-15 K=0-15, lanes 16-31 K=16-31;
        // VGPR i holds fp16 pair (k=2i, k=2i+1) of column n).
        uint4 b0 = *(const uint4*)(browbase + kb0);
        uint4 b1 = *(const uint4*)(browbase + kb0 + 4);

        V16 bop;
        unsigned int raw[8] = { b0.x, b0.y, b0.z, b0.w, b1.x, b1.y, b1.z, b1.w };
        #pragma unroll
        for (int i = 0; i < 8; ++i) {
            // nibble -> fp16 (q - 8): (0x6400 | q) is 1024+q; subtract 1032.
            unsigned int r = 0x64006400u | (raw[i] & 0xFu) | ((raw[i] & 0xF0u) << 12);
            h2 v = __builtin_bit_cast(h2, r) + bias;   // v_pk_add_f16
            bop.u[i] = __builtin_bit_cast(unsigned int, v);
        }

        // A-operand: only row M=0 is live. Matching B's k order, lane 0 holds
        // k 0-15, lane 16 holds k 16-31; all other lanes zero.
        V16 aop;
        if (col == 0) {
            aop.q[0] = *(const uint4*)&lA[g * GROUP + hi * 16];
            aop.q[1] = *(const uint4*)&lA[g * GROUP + hi * 16 + 8];
        } else {
            aop.q[0] = uint4{0u, 0u, 0u, 0u};
            aop.q[1] = uint4{0u, 0u, 0u, 0u};
        }

        // D = A x Q(g), row 0 = raw dot of (q-8) with A over this group.
        v8f d = __builtin_amdgcn_wmma_f32_16x16x32_f16(
            false, aop.h, false, bop.h, (short)0, czero, false, false);

        // Fold per-group scale/zero once: out += s * dot + z * sum(A_group).
        unsigned int szw = SZ[(long)n * NGROUP + g];
        h2 szh = __builtin_bit_cast(h2, szw);
        acc += (float)szh.x * d[0] + (float)szh.y * asum[g];
    }

    if (lane < 16) partial[wave * 16 + col] = acc;
    __syncthreads();

    if (t < 16) {
        float s = 0.f;
        #pragma unroll
        for (int w = 0; w < WAVES; ++w) s += partial[w * 16 + t];
        out[nbase + t] = s;
    }
}

extern "C" void kernel_launch(void* const* d_in, const int* in_sizes, int n_in,
                              void* d_out, int out_size, void* d_ws, size_t ws_size,
                              hipStream_t stream) {
    (void)n_in; (void)out_size; (void)d_ws; (void)ws_size;
    const _Float16*     A  = (const _Float16*)d_in[0];
    const int*          B  = (const int*)d_in[1];
    const unsigned int* SZ = (const unsigned int*)d_in[2];
    float*              o  = (float*)d_out;

    const int N = in_sizes[1] / KB;        // 16384
    dim3 grid(N / 16), block(256);
    w4a16_gemv_wmma<<<grid, block, 0, stream>>>(A, B, SZ, o);
}
